// DbrxExperts_40492951667585
// MI455X (gfx1250) — matrix operand order repaired and checked by reading, compiled, not run on hardware
//
#include <hip/hip_runtime.h>
#include <hip/hip_bf16.h>

// ---- problem constants (from reference) ----
#define TT 2048            // tokens = B*S
#define HH 2048            // hidden
#define FF 2048            // ffn dim
#define EE 8               // experts
#define KK 2               // top-k
#define MAXTOK (TT*KK)     // per-expert worst-case routed slots
#define NPAIR  (TT*KK)     // total routed pairs
#define DUMMY  NPAIR       // dummy hmid row for padded slots

// ---- tile config ----
#define MT  32             // tokens per block tile (2 WMMA M-subtiles)
#define NT  64             // N per block tile (4 WMMA N-subtiles)
#define KT2 64             // K chunk per staging step (2 x WMMA K=32)

typedef __bf16 v16bf __attribute__((ext_vector_type(16)));
typedef float  v8f   __attribute__((ext_vector_type(8)));

union FragAB { v16bf v; unsigned u[8]; };

__device__ __forceinline__ unsigned short f2bf(float f) {
    union { float f; unsigned u; } v; v.f = f;
    unsigned r = v.u + 0x7FFFu + ((v.u >> 16) & 1u);   // round-to-nearest-even
    return (unsigned short)(r >> 16);
}
__device__ __forceinline__ unsigned pack2(float lo, float hi) {
    return (unsigned)f2bf(lo) | ((unsigned)f2bf(hi) << 16);
}

#if __has_builtin(__builtin_amdgcn_rcpf)
#define FAST_RCP(x) __builtin_amdgcn_rcpf(x)
#else
#define FAST_RCP(x) (1.0f / (x))
#endif

// ---- CDNA5 async global->LDS copy (GVS form), tracked by ASYNCcnt ----
__device__ __forceinline__ unsigned lds_off32(const void* p) {
    // generic LDS pointer: aperture in bits[63:32], allocation offset in bits[31:0]
    return (unsigned)(unsigned long long)p;
}
__device__ __forceinline__ void async_to_lds_b128(unsigned lds_addr, unsigned gofs,
                                                  const void* sbase) {
    asm volatile("global_load_async_to_lds_b128 %0, %1, %2"
                 :: "v"(lds_addr), "v"(gofs), "s"(sbase) : "memory");
}
__device__ __forceinline__ void async_wait0() {
    asm volatile("s_wait_asynccnt 0x0" ::: "memory");
}

// ---- zero output + expert counters ----
__global__ void k_zero(float* __restrict__ out, int* __restrict__ cnt) {
    int i = blockIdx.x * blockDim.x + threadIdx.x;
    if (i < TT * HH) out[i] = 0.f;
    if (i < EE) cnt[i] = 0;
}

// ---- convert hidden states fp32 -> bf16 ----
__global__ void k_cvt(const float* __restrict__ x, unsigned short* __restrict__ xb) {
    int i = blockIdx.x * blockDim.x + threadIdx.x;
    if (i < TT * HH) xb[i] = f2bf(x[i]);
}

// ---- bucket routed pairs per expert ----
__global__ void k_route(const int* __restrict__ top_e, const float* __restrict__ top_w,
                        int* __restrict__ cnt, int* __restrict__ tidx,
                        float* __restrict__ tw, int* __restrict__ ps) {
    int i = blockIdx.x * blockDim.x + threadIdx.x;
    if (i >= NPAIR) return;
    int e = top_e[i];
    if (e < 0 || e >= EE) e = 0;
    int s = atomicAdd(&cnt[e], 1);
    tidx[e * MAXTOK + s] = i / KK;    // token id
    tw  [e * MAXTOK + s] = top_w[i];  // router weight
    ps  [e * MAXTOK + s] = i;         // pair slot (row in hmid)
}

// ---- stage 1: h = silu(x*Wg^T) * (x*Wu^T), per routed pair ----
__global__ __launch_bounds__(256) void k_mlp1(
    const unsigned short* __restrict__ xb,
    const float* __restrict__ wg, const float* __restrict__ wu,
    const int* __restrict__ cnt, const int* __restrict__ tidx,
    const int* __restrict__ ps, unsigned short* __restrict__ hmid)
{
    const int e    = blockIdx.y;
    const int n    = cnt[e];
    const int base = blockIdx.x * MT;
    if (base >= n) return;

    __shared__ unsigned Xs [MT][KT2/2];   // A tile: [m][k] bf16 pairs (32x64)
    __shared__ unsigned Wgs[NT][KT2/2];   // B tile transposed: [n=f][k=h]
    __shared__ unsigned Wus[NT][KT2/2];
    __shared__ int s_tok[MT];
    __shared__ int s_ps [MT];

    const int tid = threadIdx.x;
    if (tid < MT) {
        int slot = base + tid;
        if (slot < n) { s_tok[tid] = tidx[e*MAXTOK+slot]; s_ps[tid] = ps[e*MAXTOK+slot]; }
        else          { s_tok[tid] = 0;                   s_ps[tid] = DUMMY; }
    }
    __syncthreads();

    const int lane = tid & 31, wave = tid >> 5;
    const int mi = wave >> 2, ni = wave & 3;     // 2x4 subtiles of 32x64
    const int g  = lane >> 4, ln = lane & 15;

    // async X-tile copy coords: one b128 per thread (32 rows x 128B)
    const int      xr    = tid >> 3;                 // row 0..31
    const int      xc    = (tid & 7) * 4;            // uint col
    const unsigned xlds  = lds_off32(&Xs[xr][xc]);
    const unsigned xrow  = (unsigned)((size_t)s_tok[xr] * HH * 2) + (unsigned)((tid & 7) * 16);

    const float* wgE = wg + (size_t)e * FF * HH;
    const float* wuE = wu + (size_t)e * FF * HH;

    for (int ft = 0; ft < FF / NT; ++ft) {
        v8f cg = {}; v8f cu = {};
        const int fbase = ft * NT;
        for (int hk = 0; hk < HH / KT2; ++hk) {
            const int kbase = hk * KT2;
            // async-stage gathered token rows (bf16 already): global -> LDS, ASYNCcnt
            async_to_lds_b128(xlds, xrow + (unsigned)(kbase * 2), xb);
            // stage gate/up weight tiles, fp32 -> bf16, transposed [f][h]
            for (int p = tid; p < NT * (KT2/2); p += 256) {
                int r = p >> 5, c = p & 31;
                const float* sg = wgE + (size_t)(fbase + r) * HH + kbase + 2*c;
                const float* su = wuE + (size_t)(fbase + r) * HH + kbase + 2*c;
                Wgs[r][c] = pack2(sg[0], sg[1]);
                Wus[r][c] = pack2(su[0], su[1]);
            }
            __builtin_prefetch(wgE + (size_t)fbase * HH + kbase + KT2, 0, 1);
            __builtin_prefetch(wuE + (size_t)fbase * HH + kbase + KT2, 0, 1);
            async_wait0();
            __syncthreads();

            // A 16x32 bf16 frags (two K=32 chunks)
            FragAB a0, a1, bg0, bg1, bu0, bu1;
            const unsigned* Xrow = Xs[mi*16 + ln];
            #pragma unroll
            for (int v = 0; v < 8; ++v) { a0.u[v] = Xrow[8*g + v];        a1.u[v] = Xrow[16 + 8*g + v]; }
            const unsigned* Gr = Wgs[ni*16 + ln];
            const unsigned* Ur = Wus[ni*16 + ln];
            #pragma unroll
            for (int v = 0; v < 8; ++v) {
                bg0.u[v] = Gr[8*g + v];       bg1.u[v] = Gr[16 + 8*g + v];
                bu0.u[v] = Ur[8*g + v];       bu1.u[v] = Ur[16 + 8*g + v];
            }
            cg = __builtin_amdgcn_wmma_f32_16x16x32_bf16(false, a0.v, false, bg0.v, (short)0, cg, false, false);
            cg = __builtin_amdgcn_wmma_f32_16x16x32_bf16(false, a1.v, false, bg1.v, (short)0, cg, false, false);
            cu = __builtin_amdgcn_wmma_f32_16x16x32_bf16(false, a0.v, false, bu0.v, (short)0, cu, false, false);
            cu = __builtin_amdgcn_wmma_f32_16x16x32_bf16(false, a1.v, false, bu1.v, (short)0, cu, false, false);
            __syncthreads();
        }
        // epilogue: silu(g)*u -> hmid (bf16). D layout: M = r + 8*g, N = ln
        #pragma unroll
        for (int r = 0; r < 8; ++r) {
            int m    = mi*16 + r + 8*g;
            int fcol = fbase + ni*16 + ln;
            float gv = cg[r], uv = cu[r];
            float hv = gv * FAST_RCP(1.f + __expf(-gv)) * uv;
            hmid[(size_t)s_ps[m] * FF + fcol] = f2bf(hv);
        }
    }
}

// ---- stage 2: out[t] += w * (h * Wd^T) ----
__global__ __launch_bounds__(256) void k_mlp2(
    const unsigned short* __restrict__ hmid,
    const float* __restrict__ wd,
    const int* __restrict__ cnt, const int* __restrict__ tidx,
    const float* __restrict__ tw, const int* __restrict__ ps,
    float* __restrict__ out)
{
    const int e    = blockIdx.z;
    const int n    = cnt[e];
    const int base = blockIdx.x * MT;
    if (base >= n) return;
    const int hbase = blockIdx.y * NT;

    __shared__ unsigned As [MT][KT2/2];
    __shared__ unsigned Wds[NT][KT2/2];
    __shared__ int   s_tok[MT];
    __shared__ int   s_ps [MT];
    __shared__ float s_w  [MT];

    const int tid = threadIdx.x;
    if (tid < MT) {
        int slot = base + tid;
        if (slot < n) { s_tok[tid]=tidx[e*MAXTOK+slot]; s_ps[tid]=ps[e*MAXTOK+slot]; s_w[tid]=tw[e*MAXTOK+slot]; }
        else          { s_tok[tid]=0;                   s_ps[tid]=DUMMY;             s_w[tid]=0.f; }
    }
    __syncthreads();

    const int lane = tid & 31, wave = tid >> 5;
    const int mi = wave >> 2, ni = wave & 3;
    const int g  = lane >> 4, ln = lane & 15;

    const int      xr   = tid >> 3;
    const int      xc   = (tid & 7) * 4;
    const unsigned xlds = lds_off32(&As[xr][xc]);
    const unsigned xrow = (unsigned)((size_t)s_ps[xr] * FF * 2) + (unsigned)((tid & 7) * 16);

    const float* wdE = wd + (size_t)e * HH * FF;

    v8f acc = {};
    for (int fk = 0; fk < FF / KT2; ++fk) {
        const int kbase = fk * KT2;
        // async-stage intermediate rows (bf16 already)
        async_to_lds_b128(xlds, xrow + (unsigned)(kbase * 2), hmid);
        for (int p = tid; p < NT * (KT2/2); p += 256) {
            int r = p >> 5, c = p & 31;
            const float* sd = wdE + (size_t)(hbase + r) * FF + kbase + 2*c;
            Wds[r][c] = pack2(sd[0], sd[1]);
        }
        __builtin_prefetch(wdE + (size_t)hbase * FF + kbase + KT2, 0, 1);
        async_wait0();
        __syncthreads();

        FragAB a0, a1, b0, b1;
        const unsigned* Ar = As[mi*16 + ln];
        #pragma unroll
        for (int v = 0; v < 8; ++v) { a0.u[v] = Ar[8*g + v];       a1.u[v] = Ar[16 + 8*g + v]; }
        const unsigned* Dr = Wds[ni*16 + ln];
        #pragma unroll
        for (int v = 0; v < 8; ++v) { b0.u[v] = Dr[8*g + v]; b1.u[v] = Dr[16 + 8*g + v]; }

        acc = __builtin_amdgcn_wmma_f32_16x16x32_bf16(false, a0.v, false, b0.v, (short)0, acc, false, false);
        acc = __builtin_amdgcn_wmma_f32_16x16x32_bf16(false, a1.v, false, b1.v, (short)0, acc, false, false);
        __syncthreads();
    }
    // scale by router weight, scatter-add into output (token may hit 2 experts)
    #pragma unroll
    for (int r = 0; r < 8; ++r) {
        int m    = mi*16 + r + 8*g;
        int hcol = hbase + ni*16 + ln;
        atomicAdd(&out[(size_t)s_tok[m] * HH + hcol], acc[r] * s_w[m]);
    }
}

extern "C" void kernel_launch(void* const* d_in, const int* in_sizes, int n_in,
                              void* d_out, int out_size, void* d_ws, size_t ws_size,
                              hipStream_t stream)
{
    (void)in_sizes; (void)n_in; (void)out_size; (void)ws_size;
    const float* x     = (const float*)d_in[0];   // hidden_states [B,S,H]
    const float* tw_in = (const float*)d_in[1];   // top_weights   [T,K]
    const float* wg    = (const float*)d_in[2];   // w_gate [E,F,H]
    const float* wu    = (const float*)d_in[3];   // w_up   [E,F,H]
    const float* wd    = (const float*)d_in[4];   // w_down [E,H,F]
    const int*   te    = (const int*)d_in[5];     // top_experts   [T,K]
    float* out = (float*)d_out;

    char* ws = (char*)d_ws; size_t off = 0;
    auto alloc = [&](size_t bytes) -> void* {
        void* p = ws + off; off = (off + bytes + 255) & ~(size_t)255; return p;
    };
    int*   cnt  = (int*)  alloc(EE * sizeof(int));
    int*   tidx = (int*)  alloc((size_t)EE * MAXTOK * sizeof(int));
    float* twb  = (float*)alloc((size_t)EE * MAXTOK * sizeof(float));
    int*   psb  = (int*)  alloc((size_t)EE * MAXTOK * sizeof(int));
    unsigned short* xb   = (unsigned short*)alloc((size_t)TT * HH * sizeof(unsigned short));
    unsigned short* hmid = (unsigned short*)alloc(((size_t)NPAIR + MT) * FF * sizeof(unsigned short));

    k_zero <<<(TT*HH + 255)/256, 256, 0, stream>>>(out, cnt);
    k_cvt  <<<(TT*HH + 255)/256, 256, 0, stream>>>(x, xb);
    k_route<<<(NPAIR + 255)/256, 256, 0, stream>>>(te, tw_in, cnt, tidx, twb, psb);
    k_mlp1 <<<dim3(MAXTOK/MT, EE),        256, 0, stream>>>(xb, wg, wu, cnt, tidx, psb, hmid);
    k_mlp2 <<<dim3(MAXTOK/MT, HH/NT, EE), 256, 0, stream>>>(hmid, wd, cnt, tidx, twb, psb, out);
}
